// CausalSelfAttention_3272765079739
// MI455X (gfx1250) — matrix operand order repaired and checked by reading, not compiled, not run on hardware
//
#include <hip/hip_runtime.h>
#include <hip/hip_bf16.h>

#define B_ 2
#define T_ 2048
#define C_ 1024
#define H_ 16
#define HD 64

typedef __bf16 bf16_t;
typedef __attribute__((ext_vector_type(16))) __bf16 v16bf;
typedef __attribute__((ext_vector_type(8)))  float  v8f;

union frag_pair { uint4 q[2]; v16bf v; };

__device__ __forceinline__ v16bf load_frag2(const bf16_t* p0, const bf16_t* p1) {
  frag_pair u;
  u.q[0] = *reinterpret_cast<const uint4*>(p0);
  u.q[1] = *reinterpret_cast<const uint4*>(p1);
  return u.v;
}

// 16B global -> LDS async copy (CDNA5 GLOBAL_LOAD_ASYNC_TO_LDS_B128, GV mode).
// lds_off: wave-relative LDS byte address (= low 32 bits of generic shared ptr).
__device__ __forceinline__ void async_ld_b128(unsigned lds_off, const bf16_t* g) {
  asm volatile("global_load_async_to_lds_b128 %0, %1, off"
               :: "v"(lds_off), "v"(g)
               : "memory");
}

__device__ __forceinline__ unsigned lds_addr(const void* p) {
  return (unsigned)(uintptr_t)p;
}

// ---------------- prep kernels ----------------

__global__ void convert_f32_bf16(const float* __restrict__ in, bf16_t* __restrict__ out, int n) {
  int i = blockIdx.x * blockDim.x + threadIdx.x;
  if (i < n) out[i] = (bf16_t)in[i];
}

// in[K][N] (fp32) -> out[N][K] (bf16)
__global__ void transpose_f32_bf16(const float* __restrict__ in, bf16_t* __restrict__ out,
                                   int K, int N, int total) {
  int i = blockIdx.x * blockDim.x + threadIdx.x;
  if (i < total) {
    int kk = i / N;
    int nn = i - kk * N;
    out[(size_t)nn * K + kk] = (bf16_t)in[i];
  }
}

// ---------------- bf16 WMMA GEMM (async double-buffered) ----------------
// C[M,N] = A[M,K] * Bt[N,K]^T + bias.  Block tile 128x128, 8 waves of 64x32.
// mode 0: write fp32 to outF[M,N].
// mode 1: QKV epilogue: scatter into q/k/v [B,H,T,HD] bf16, Q scaled by 1/8.
__global__ __launch_bounds__(256) void gemm_bf16_wmma(
    const bf16_t* __restrict__ A,
    const bf16_t* __restrict__ Bt,
    const float*  __restrict__ bias,
    int M, int N, int K, int mode,
    float* __restrict__ outF,
    bf16_t* __restrict__ qOut, bf16_t* __restrict__ kOut, bf16_t* __restrict__ vOut)
{
  __shared__ __align__(16) bf16_t lds_a[2][128 * 40];
  __shared__ __align__(16) bf16_t lds_b[2][128 * 40];

  const int tid  = threadIdx.x;
  const int lane = tid & 31;
  const int w    = tid >> 5;
  const int l15  = lane & 15;
  const int half = lane >> 4;

  const int m0   = blockIdx.y * 128;
  const int n0   = blockIdx.x * 128;
  const int wrow = (w >> 2) * 64;
  const int wcol = (w & 3) * 32;

  v8f acc[4][2];
  #pragma unroll
  for (int mi = 0; mi < 4; ++mi)
    #pragma unroll
    for (int ni = 0; ni < 2; ++ni)
      #pragma unroll
      for (int i = 0; i < 8; ++i) acc[mi][ni][i] = 0.0f;

  const int row2 = tid >> 1;   // 0..127
  const int hh   = tid & 1;    // which 16-element half of the 32-wide K slab

  const bf16_t* gA = A  + (size_t)(m0 + row2) * K + hh * 16;
  const bf16_t* gB = Bt + (size_t)(n0 + row2) * K + hh * 16;

  auto stage = [&](int k0, int buf) {
    unsigned la = lds_addr(&lds_a[buf][row2 * 40 + hh * 16]);
    async_ld_b128(la,      gA + k0);
    async_ld_b128(la + 16, gA + k0 + 8);
    unsigned lb = lds_addr(&lds_b[buf][row2 * 40 + hh * 16]);
    async_ld_b128(lb,      gB + k0);
    async_ld_b128(lb + 16, gB + k0 + 8);
  };

  const int nT = K >> 5;
  stage(0, 0);
  for (int t = 0; t < nT; ++t) {
    const int buf = t & 1;
    if (t + 1 < nT) {
      stage((t + 1) << 5, buf ^ 1);
      asm volatile("s_wait_asynccnt 0x4" ::: "memory");  // older 4 (this tile) done
    } else {
      asm volatile("s_wait_asynccnt 0x0" ::: "memory");
    }
    __syncthreads();

    v16bf af[4];
    #pragma unroll
    for (int mi = 0; mi < 4; ++mi) {
      const bf16_t* p = &lds_a[buf][(wrow + mi * 16 + l15) * 40 + half * 8];
      af[mi] = load_frag2(p, p + 16);              // K runs {8kg..}, {16+8kg..}
    }
    v16bf bfr[2];
    #pragma unroll
    for (int ni = 0; ni < 2; ++ni) {
      const bf16_t* p = &lds_b[buf][(wcol + ni * 16 + l15) * 40 + half * 8];
      bfr[ni] = load_frag2(p, p + 16);             // K runs {8kg..}, {16+8kg..}
    }
    #pragma unroll
    for (int mi = 0; mi < 4; ++mi)
      #pragma unroll
      for (int ni = 0; ni < 2; ++ni)
        acc[mi][ni] = __builtin_amdgcn_wmma_f32_16x16x32_bf16(
            false, af[mi], false, bfr[ni], (short)0, acc[mi][ni], false, false);
    __syncthreads();   // gates overwrite of this buffer by next iteration's stage
  }

  #pragma unroll
  for (int mi = 0; mi < 4; ++mi) {
    #pragma unroll
    for (int ni = 0; ni < 2; ++ni) {
      const int gcol = n0 + wcol + ni * 16 + l15;
      const float bv = bias[gcol];
      #pragma unroll
      for (int r = 0; r < 8; ++r) {
        const int grow = m0 + wrow + mi * 16 + r + half * 8;
        const float val = acc[mi][ni][r] + bv;
        if (mode == 0) {
          outF[(size_t)grow * N + gcol] = val;
        } else {
          const int which = gcol >> 10;        // / C_
          const int c     = gcol & (C_ - 1);
          const int h     = c >> 6;
          const int d     = c & 63;
          const int bi    = grow >> 11;        // / T_
          const int t     = grow & (T_ - 1);
          const size_t idx = (((size_t)bi * H_ + h) * T_ + t) * HD + d;
          if (which == 0)      qOut[idx] = (bf16_t)(val * 0.125f);  // 1/sqrt(64)
          else if (which == 1) kOut[idx] = (bf16_t)val;
          else                 vOut[idx] = (bf16_t)val;
        }
      }
    }
  }
}

// ---------------- flash attention ----------------
// grid: (T/128, B*H); block 256 (8 waves). Wave w owns query rows [q0+16w, q0+16w+16).
__global__ __launch_bounds__(256) void flash_attn_wmma(
    const bf16_t* __restrict__ Q,
    const bf16_t* __restrict__ Kb,
    const bf16_t* __restrict__ Vb,
    bf16_t* __restrict__ Y)
{
  __shared__ __align__(16) bf16_t lds_k [64 * 72];     // [key][d]
  __shared__ __align__(16) bf16_t lds_vt[64 * 72];     // [d][key]
  __shared__ __align__(16) bf16_t lds_p [8][16 * 72];  // per-wave P tile

  const int tid  = threadIdx.x;
  const int lane = tid & 31;
  const int w    = tid >> 5;
  const int l15  = lane & 15;
  const int half = lane >> 4;

  const int qt = blockIdx.x;
  const int bh = blockIdx.y;
  const int q0 = qt * 128;
  const size_t head_base = (size_t)bh * T_ * HD;

  v16bf qf[2];
  {
    const int row = q0 + w * 16 + l15;
    const bf16_t* qp = Q + head_base + (size_t)row * HD;
    #pragma unroll
    for (int c = 0; c < 2; ++c)
      qf[c] = load_frag2(qp + c * 32 + half * 8, qp + c * 32 + half * 8 + 16);
  }

  float m_r[8], l_r[8];
  v8f oacc[4];
  #pragma unroll
  for (int r = 0; r < 8; ++r) { m_r[r] = -3.0e38f; l_r[r] = 0.0f; }
  #pragma unroll
  for (int di = 0; di < 4; ++di)
    #pragma unroll
    for (int r = 0; r < 8; ++r) oacc[di][r] = 0.0f;

  const int srow = tid >> 2;   // key row 0..63
  const int sq   = tid & 3;    // 16-element quarter of d

  const int nkt = 2 * qt + 2;  // 64-key tiles covering [0, q0+128)
  for (int kt = 0; kt < nkt; ++kt) {
    const int k0 = kt * 64;
    {
      // K tile via CDNA5 async global->LDS
      const bf16_t* gk = Kb + head_base + (size_t)(k0 + srow) * HD + sq * 16;
      unsigned lk = lds_addr(&lds_k[srow * 72 + sq * 16]);
      async_ld_b128(lk,      gk);
      async_ld_b128(lk + 16, gk + 8);
      // V tile transposed (scatter, manual): lds_vt[d][key]
      const bf16_t* gv = Vb + head_base + (size_t)(k0 + srow) * HD + sq * 16;
      #pragma unroll
      for (int j = 0; j < 16; ++j)
        lds_vt[(sq * 16 + j) * 72 + srow] = gv[j];
    }
    asm volatile("s_wait_asynccnt 0x0" ::: "memory");
    __syncthreads();

    // S(16x64) = Q * K^T
    v8f s[4];
    #pragma unroll
    for (int ni = 0; ni < 4; ++ni) {
      const bf16_t* kp = &lds_k[(ni * 16 + l15) * 72 + half * 8];
      v16bf kf0 = load_frag2(kp, kp + 16);        // d 0..31
      v16bf kf1 = load_frag2(kp + 32, kp + 48);   // d 32..63
      v8f z;
      #pragma unroll
      for (int i = 0; i < 8; ++i) z[i] = 0.0f;
      s[ni] = __builtin_amdgcn_wmma_f32_16x16x32_bf16(false, qf[0], false, kf0, (short)0, z,     false, false);
      s[ni] = __builtin_amdgcn_wmma_f32_16x16x32_bf16(false, qf[1], false, kf1, (short)0, s[ni], false, false);
    }

    if (k0 + 63 > q0) {  // only tiles that can cross the diagonal
      #pragma unroll
      for (int ni = 0; ni < 4; ++ni) {
        const int key = k0 + ni * 16 + l15;
        #pragma unroll
        for (int r = 0; r < 8; ++r) {
          const int qr = q0 + w * 16 + r + half * 8;
          if (key > qr) s[ni][r] = -3.0e38f;
        }
      }
    }

    // online softmax (row stats live replicated across each 16-lane half)
    float mnew[8];
    #pragma unroll
    for (int r = 0; r < 8; ++r) {
      float mv = m_r[r];
      #pragma unroll
      for (int ni = 0; ni < 4; ++ni) mv = fmaxf(mv, s[ni][r]);
      mnew[r] = mv;
    }
    #pragma unroll
    for (int off = 8; off >= 1; off >>= 1)
      #pragma unroll
      for (int r = 0; r < 8; ++r)
        mnew[r] = fmaxf(mnew[r], __shfl_xor(mnew[r], off, 32));

    float scl[8], lsum[8];
    #pragma unroll
    for (int r = 0; r < 8; ++r) {
      scl[r] = __expf(m_r[r] - mnew[r]);
      lsum[r] = 0.0f;
      m_r[r] = mnew[r];
    }

    bf16_t* pw = &lds_p[w][0];
    #pragma unroll
    for (int ni = 0; ni < 4; ++ni) {
      #pragma unroll
      for (int r = 0; r < 8; ++r) {
        const float p = __expf(s[ni][r] - mnew[r]);
        lsum[r] += p;
        pw[(r + half * 8) * 72 + ni * 16 + l15] = (bf16_t)p;
      }
    }
    #pragma unroll
    for (int off = 8; off >= 1; off >>= 1)
      #pragma unroll
      for (int r = 0; r < 8; ++r)
        lsum[r] += __shfl_xor(lsum[r], off, 32);
    #pragma unroll
    for (int r = 0; r < 8; ++r) l_r[r] = l_r[r] * scl[r] + lsum[r];
    #pragma unroll
    for (int di = 0; di < 4; ++di)
      #pragma unroll
      for (int r = 0; r < 8; ++r)
        oacc[di][r] *= scl[r];

    // same-wave LDS RAW: P stores must land before fragment reloads
    asm volatile("s_wait_dscnt 0" ::: "memory");

    v16bf pf[2];
    #pragma unroll
    for (int kc = 0; kc < 2; ++kc) {
      const bf16_t* pp = pw + l15 * 72 + kc * 32 + half * 8;
      pf[kc] = load_frag2(pp, pp + 16);
    }
    #pragma unroll
    for (int di = 0; di < 4; ++di) {
      #pragma unroll
      for (int kc = 0; kc < 2; ++kc) {
        const bf16_t* vp = &lds_vt[(di * 16 + l15) * 72 + kc * 32 + half * 8];
        v16bf vf = load_frag2(vp, vp + 16);
        oacc[di] = __builtin_amdgcn_wmma_f32_16x16x32_bf16(
            false, pf[kc], false, vf, (short)0, oacc[di], false, false);
      }
    }
    __syncthreads();
  }

  const int b_i = bh / H_;
  const int h_i = bh - b_i * H_;
  #pragma unroll
  for (int di = 0; di < 4; ++di) {
    #pragma unroll
    for (int r = 0; r < 8; ++r) {
      const int t = q0 + w * 16 + r + half * 8;
      const int d = di * 16 + l15;
      const float val = oacc[di][r] / l_r[r];
      Y[((size_t)(b_i * T_ + t)) * C_ + h_i * HD + d] = (bf16_t)val;
    }
  }
}

// ---------------- launch ----------------

extern "C" void kernel_launch(void* const* d_in, const int* in_sizes, int n_in,
                              void* d_out, int out_size, void* d_ws, size_t ws_size,
                              hipStream_t stream) {
  (void)in_sizes; (void)n_in; (void)out_size; (void)ws_size;
  const float* x      = (const float*)d_in[0];
  const float* W_attn = (const float*)d_in[1];
  const float* b_attn = (const float*)d_in[2];
  const float* W_proj = (const float*)d_in[3];
  const float* b_proj = (const float*)d_in[4];
  float* out = (float*)d_out;

  char* ws = (char*)d_ws;
  const size_t ACT = (size_t)B_ * T_ * C_;       // 4,194,304 elements
  bf16_t* xb     = (bf16_t*)(ws);                                   //  8 MB
  bf16_t* wtAttn = (bf16_t*)(ws + ACT * 2);                         //  6 MB
  bf16_t* wtProj = (bf16_t*)(ws + ACT * 2 + (size_t)3 * C_ * C_ * 2);
  bf16_t* qB     = (bf16_t*)(ws + 16777216);
  bf16_t* kB     = (bf16_t*)(ws + 25165824);
  bf16_t* vB     = (bf16_t*)(ws + 33554432);
  bf16_t* yB     = (bf16_t*)(ws + 41943040);     // end: 50,331,648 bytes

  {
    int n = (int)ACT;
    convert_f32_bf16<<<(n + 255) / 256, 256, 0, stream>>>(x, xb, n);
  }
  {
    int n = C_ * 3 * C_;
    transpose_f32_bf16<<<(n + 255) / 256, 256, 0, stream>>>(W_attn, wtAttn, C_, 3 * C_, n);
  }
  {
    int n = C_ * C_;
    transpose_f32_bf16<<<(n + 255) / 256, 256, 0, stream>>>(W_proj, wtProj, C_, C_, n);
  }

  // QKV projection: [4096,1024] x [1024,3072]
  gemm_bf16_wmma<<<dim3(24, 32), 256, 0, stream>>>(
      xb, wtAttn, b_attn, B_ * T_, 3 * C_, C_, 1, nullptr, qB, kB, vB);

  // causal flash attention
  flash_attn_wmma<<<dim3(T_ / 128, B_ * H_), 256, 0, stream>>>(qB, kB, vB, yB);

  // output projection: [4096,1024] x [1024,1024] -> fp32 out
  gemm_bf16_wmma<<<dim3(8, 32), 256, 0, stream>>>(
      yB, wtProj, b_proj, B_ * T_, C_, C_, 0, out, nullptr, nullptr, nullptr);
}
